// KTModel_36249523978514
// MI455X (gfx1250) — matrix operand order repaired and checked by reading, not compiled, not run on hardware
//
#include <hip/hip_runtime.h>
#include <hip/hip_bf16.h>

typedef __attribute__((ext_vector_type(16))) _Float16 v16h;
typedef __attribute__((ext_vector_type(8)))  _Float16 v8h;
typedef __attribute__((ext_vector_type(8)))  float    v8f;

#define BSZ   256
#define SLEN  200
#define EMBD  256
#define K2    512   // 2*EMB

__device__ __forceinline__ float sigf(float x) {
    return 1.0f / (1.0f + __expf(-x));
}

// A-fragment slot permutation within a 32-wide k chunk (swap bits 3 and 4).
__device__ __forceinline__ int slot32(int q) {
    return (q & 7) | ((q & 8) << 1) | ((q & 16) >> 1);
}

// Load 8 consecutive f32 (32B aligned) into f16 fragment slots [s, s+8)
__device__ __forceinline__ void load8(const float* __restrict__ p, v16h& a, int s) {
    float4 u = *(const float4*)(p);
    float4 v = *(const float4*)(p + 4);
    a[s + 0] = (_Float16)u.x; a[s + 1] = (_Float16)u.y;
    a[s + 2] = (_Float16)u.z; a[s + 3] = (_Float16)u.w;
    a[s + 4] = (_Float16)v.x; a[s + 5] = (_Float16)v.y;
    a[s + 6] = (_Float16)v.z; a[s + 7] = (_Float16)v.w;
}

// ---------------------------------------------------------------------------
// Prep: repack fc_w / fc2_w (EMB x 2EMB f32) into f16 B-fragment layout:
//   index = ((kc*16 + nt)*32 + lane)*16 + j
//   n = nt*16 + (lane&15);  k = kc*32 + (lane>=16 ? 8 : 0) + (j&7) + (j>=8 ? 16 : 0)
// ---------------------------------------------------------------------------
__global__ void prep_weights(const float* __restrict__ fc_w,
                             const float* __restrict__ fc2_w,
                             _Float16* __restrict__ Wf1h,
                             _Float16* __restrict__ Wf2h) {
    int idx   = blockIdx.x * blockDim.x + threadIdx.x;   // [0, 262144)
    int which = idx >> 17;
    int r     = idx & 131071;
    int j     = r & 15;
    int lane  = (r >> 4) & 31;
    int nt    = (r >> 9) & 15;
    int kc    = r >> 13;
    int n     = nt * 16 + (lane & 15);
    int k     = kc * 32 + ((lane >= 16) ? 8 : 0) + (j & 7) + ((j & 8) << 1);
    const float* W = which ? fc2_w : fc_w;
    _Float16* D    = which ? Wf2h  : Wf1h;
    D[r] = (_Float16)W[n * K2 + k];
}

// ---------------------------------------------------------------------------
// Phase 1: x = [E_skill|E_diff] @ fc_w.T + fc_b over 51200 rows.
// Each wave computes a 16x32 tile (two N-tiles share the A gather).
// x stored ONLY as f16 in A-fragment (slot-permuted) order.
// ---------------------------------------------------------------------------
__global__ void __launch_bounds__(128, 1)
phase1_gemm(const int* __restrict__ skill, const int* __restrict__ diff,
            const float* __restrict__ Eskill, const float* __restrict__ Ediff,
            const _Float16* __restrict__ Wf1h, const float* __restrict__ fcb,
            _Float16* __restrict__ xF16) {
    int wave   = threadIdx.x >> 5;
    int lane   = threadIdx.x & 31;
    int tileId = blockIdx.x * 4 + wave;        // 25600 tiles (3200 M x 8 Npair)
    int mt  = tileId >> 3;
    int nt0 = (tileId & 7) * 2;
    int nt1 = nt0 + 1;
    int m   = lane & 15;
    bool hi = lane >= 16;

    int row = mt * 16 + m;
    const float* rs = Eskill + (size_t)skill[row] * EMBD;
    const float* rd = Ediff  + (size_t)diff[row]  * EMBD;

    v8f acc0 = {}, acc1 = {};
#pragma unroll
    for (int kc = 0; kc < 16; ++kc) {
        v16h a;
        int g0 = kc * 32 + (hi ? 8 : 0);
        int g1 = g0 + 16;
        load8(g0 < EMBD ? rs + g0 : rd + (g0 - EMBD), a, 0);
        load8(g1 < EMBD ? rs + g1 : rd + (g1 - EMBD), a, 8);
        v16h b0 = *(const v16h*)(Wf1h + ((size_t)((kc * 16 + nt0) * 32 + lane)) * 16);
        v16h b1 = *(const v16h*)(Wf1h + ((size_t)((kc * 16 + nt1) * 32 + lane)) * 16);
        acc0 = __builtin_amdgcn_wmma_f32_16x16x32_f16(
            false, a, false, b0, (short)0, acc0, false, false);
        acc1 = __builtin_amdgcn_wmma_f32_16x16x32_f16(
            false, a, false, b1, (short)0, acc1, false, false);
    }

    int n0 = nt0 * 16 + m;
    int n1 = nt1 * 16 + m;
    float bias0 = fcb[n0], bias1 = fcb[n1];
    int nPos0 = (n0 & ~31) | slot32(n0 & 31);
    int nPos1 = (n1 & ~31) | slot32(n1 & 31);
#pragma unroll
    for (int r = 0; r < 8; ++r) {
        size_t rr = (size_t)(mt * 16 + r + (hi ? 8 : 0)) * EMBD;
        xF16[rr + nPos0] = (_Float16)(acc0[r] + bias0);
        xF16[rr + nPos1] = (_Float16)(acc1[r] + bias1);
    }
}

// ---------------------------------------------------------------------------
// Phase 2: recurrent scan. 16 WGs x 16 samples; 256 threads (8 waves).
// fc2_w cached in LDS; h staged as f16 fragments; x_t read as pre-packed
// f16 fragments. y-dot is deferred: h_new is streamed to hOut (permuted
// order) and reduced by the ydot kernel. 3 barriers per step, no cvt,
// no reductions on the serial critical path.
// ---------------------------------------------------------------------------
__global__ void __launch_bounds__(256, 1)
phase2_scan(const _Float16* __restrict__ xF16,
            const _Float16* __restrict__ Wf2h, const float* __restrict__ fc2b,
            const int* __restrict__ gap, const int* __restrict__ time_used,
            const int* __restrict__ hints, const int* __restrict__ answer,
            const float* __restrict__ Egap, const float* __restrict__ Etime,
            const float* __restrict__ Ehints, const float* __restrict__ Eans,
            const float* __restrict__ Wq1, const float* __restrict__ Wk1,
            const float* __restrict__ Wv1, const float* __restrict__ Wq2,
            const float* __restrict__ Wk2, const float* __restrict__ Wv2,
            const float* __restrict__ Wq3, const float* __restrict__ Wk3,
            const float* __restrict__ Wv3,
            float* __restrict__ hOut) {
    __shared__ _Float16 Wf2LDS[16 * 512 * 16];   // 131072 halves = 256 KB
    __shared__ _Float16 hBufH[16 * EMBD];        // 8 KB, A-fragment order
    __shared__ float    xtBuf[16 * EMBD];        // 16 KB
    __shared__ int      idxBuf[2][64];           // double-buffered indices

    const int tid  = threadIdx.x;
    const int b0   = blockIdx.x * 16;
    const int lane = tid & 31;
    const int wave = tid >> 5;
    const int m    = lane & 15;
    const bool hi  = lane >= 16;
    const int e    = tid;                        // owned column
    const int ePos = (e & ~31) | slot32(e & 31); // fragment slot of column e

    // cooperative copy of fc2_w fragments into LDS (b128 both sides)
    {
        const float4* src = (const float4*)Wf2h;
        float4* dst = (float4*)Wf2LDS;
#pragma unroll
        for (int i = 0; i < 64; ++i)
            dst[tid + i * 256] = src[tid + i * 256];
    }

    float h[16], c[16];
#pragma unroll
    for (int s = 0; s < 16; ++s) { h[s] = 0.0f; c[s] = 0.0f; }

    if (tid < 64) {                              // stage indices for t=0
        int s = tid & 15, which = tid >> 4;
        const int* p = which == 0 ? gap : which == 1 ? time_used
                     : which == 2 ? hints : answer;
        idxBuf[0][tid] = p[(b0 + s) * SLEN];
    }
    const float bias_e = fc2b[e];
    __syncthreads();

    for (int t = 0; t < SLEN - 1; ++t) {
        const int* cur = idxBuf[t & 1];

        // ---- step 1: stage next indices; h -= sig(e_gap); stage h f16 ----
        if (tid < 64) {
            int s = tid & 15, which = tid >> 4;
            const int* p = which == 0 ? gap : which == 1 ? time_used
                         : which == 2 ? hints : answer;
            idxBuf[(t + 1) & 1][tid] = p[(b0 + s) * SLEN + t + 1];
        }
#pragma unroll
        for (int s = 0; s < 16; ++s) {
            h[s] -= sigf(Egap[(size_t)cur[s] * EMBD + e]);
            hBufH[s * EMBD + ePos] = (_Float16)h[s];
        }
        __syncthreads();

        // ---- step 2: xt = [x_t | h] @ fc2_w.T (M=16, N=256, K=512) ----
        {
            const _Float16* xr = xF16 + ((size_t)(b0 + m) * SLEN + t) * EMBD;
            const _Float16* hr = &hBufH[m * EMBD];
            const int nt0 = wave * 2, nt1 = nt0 + 1;
            v8f acc0 = {}, acc1 = {};
#pragma unroll
            for (int kc = 0; kc < 16; ++kc) {
                int off = kc * 32 + (hi ? 16 : 0);
                v16h a = (kc < 8) ? *(const v16h*)(xr + off)
                                  : *(const v16h*)(hr + (off - EMBD));
                v16h bf0 = *(const v16h*)(&Wf2LDS[((kc * 16 + nt0) * 32 + lane) * 16]);
                v16h bf1 = *(const v16h*)(&Wf2LDS[((kc * 16 + nt1) * 32 + lane) * 16]);
                acc0 = __builtin_amdgcn_wmma_f32_16x16x32_f16(
                    false, a, false, bf0, (short)0, acc0, false, false);
                acc1 = __builtin_amdgcn_wmma_f32_16x16x32_f16(
                    false, a, false, bf1, (short)0, acc1, false, false);
            }
            int n0 = nt0 * 16 + m, n1 = nt1 * 16 + m;
#pragma unroll
            for (int r = 0; r < 8; ++r) {
                int mr = r + (hi ? 8 : 0);
                xtBuf[mr * EMBD + n0] = acc0[r];
                xtBuf[mr * EMBD + n1] = acc1[r];
            }
        }
        __syncthreads();

        // ---- step 3: gate math; stream h_new to hOut (permuted order) ----
#pragma unroll
        for (int s = 0; s < 16; ++s) {
            const size_t we = (size_t)(b0 + s) * EMBD + e;
            float xt = xtBuf[s * EMBD + e] + bias_e;
            float u  = Etime [(size_t)cur[16 + s] * EMBD + e];
            float hn = Ehints[(size_t)cur[32 + s] * EMBD + e];
            float av = Eans  [(size_t)cur[48 + s] * EMBD + e];

            float s1 = sigf((Wq1[we] * xt) * (Wk1[we] * c[s]) * 0.0625f);
            float o1 = (Wv1[we] * u) * s1;
            float s2 = sigf((Wq2[we] * xt) * (Wk2[we] * c[s]) * 0.0625f);
            float o2 = (Wv2[we] * hn) * s2;
            float ov = o1 + o2;
            c[s] = tanhf(ov) * sigf(ov) + c[s];

            float s3 = sigf((Wk3[we] * xt) * (Wq3[we] * h[s]) * 0.0625f);
            float o3 = (Wv3[we] * av) * s3;
            h[s] = tanhf(o3) * sigf(o3) + c[s];

            hOut[((size_t)(b0 + s) * (SLEN - 1) + t) * EMBD + ePos] = h[s];
        }
        __syncthreads();
    }
}

// ---------------------------------------------------------------------------
// ydot: out[b, l] = l==0 ? 0 : sigmoid( dot(x[b, l], h[b, l-1]) ).
// Both xF16 and hOut are stored in the same slot-permuted order, so a
// straight elementwise dot is correct. One wave per output element.
// ---------------------------------------------------------------------------
__global__ void __launch_bounds__(256, 1)
ydot(const _Float16* __restrict__ xF16, const float* __restrict__ hOut,
     float* __restrict__ out) {
    int wid  = (blockIdx.x * blockDim.x + threadIdx.x) >> 5;  // [0, 51200)
    int lane = threadIdx.x & 31;
    int b = wid / SLEN, l = wid - b * SLEN;
    if (l == 0) {
        if (lane == 0) out[(size_t)b * SLEN] = 0.0f;
        return;
    }
    const _Float16* xr = xF16 + ((size_t)b * SLEN + l) * EMBD + lane * 8;
    const float*    hr = hOut + ((size_t)b * (SLEN - 1) + (l - 1)) * EMBD + lane * 8;
    v8h xv = *(const v8h*)xr;
    float4 h0 = *(const float4*)(hr);
    float4 h1 = *(const float4*)(hr + 4);
    float sum = (float)xv[0] * h0.x + (float)xv[1] * h0.y +
                (float)xv[2] * h0.z + (float)xv[3] * h0.w +
                (float)xv[4] * h1.x + (float)xv[5] * h1.y +
                (float)xv[6] * h1.z + (float)xv[7] * h1.w;
#pragma unroll
    for (int off = 16; off > 0; off >>= 1)
        sum += __shfl_xor(sum, off, 32);
    if (lane == 0) out[(size_t)b * SLEN + l] = sigf(sum);
}

extern "C" void kernel_launch(void* const* d_in, const int* in_sizes, int n_in,
                              void* d_out, int out_size, void* d_ws, size_t ws_size,
                              hipStream_t stream) {
    const int* skill     = (const int*)d_in[0];
    const int* answer    = (const int*)d_in[1];
    const int* diff      = (const int*)d_in[2];
    const int* hints     = (const int*)d_in[3];
    const int* gap       = (const int*)d_in[4];
    const int* time_used = (const int*)d_in[5];
    const float* E_skill = (const float*)d_in[6];
    const float* E_answer= (const float*)d_in[7];
    const float* E_diff  = (const float*)d_in[8];
    const float* E_hints = (const float*)d_in[9];
    const float* E_gap   = (const float*)d_in[10];
    const float* E_time  = (const float*)d_in[11];
    const float* Wq1 = (const float*)d_in[12];
    const float* Wk1 = (const float*)d_in[13];
    const float* Wv1 = (const float*)d_in[14];
    const float* Wq2 = (const float*)d_in[15];
    const float* Wk2 = (const float*)d_in[16];
    const float* Wv2 = (const float*)d_in[17];
    const float* Wq3 = (const float*)d_in[18];
    const float* Wk3 = (const float*)d_in[19];
    const float* Wv3 = (const float*)d_in[20];
    const float* fc_w  = (const float*)d_in[21];
    const float* fc_b  = (const float*)d_in[22];
    const float* fc2_w = (const float*)d_in[23];
    const float* fc2_b = (const float*)d_in[24];
    float* out = (float*)d_out;

    // Workspace layout
    _Float16* Wf1h = (_Float16*)d_ws;                             // 262144 B
    _Float16* Wf2h = Wf1h + 131072;                                // 262144 B
    _Float16* xF16 = (_Float16*)((char*)d_ws + 524288);            // 26214400 B
    float*    hOut = (float*)((char*)d_ws + 524288 + 26214400);    // 52166656 B

    prep_weights<<<1024, 256, 0, stream>>>(fc_w, fc2_w, Wf1h, Wf2h);

    phase1_gemm<<<6400, 128, 0, stream>>>(skill, diff, E_skill, E_diff,
                                          Wf1h, fc_b, xF16);

    phase2_scan<<<16, 256, 0, stream>>>(xF16, Wf2h, fc2_b,
                                        gap, time_used, hints, answer,
                                        E_gap, E_time, E_hints, E_answer,
                                        Wq1, Wk1, Wv1, Wq2, Wk2, Wv2,
                                        Wq3, Wk3, Wv3, hOut);

    ydot<<<6400, 256, 0, stream>>>(xF16, hOut, out);
}
